// Allegro_Module_30623116821359
// MI455X (gfx1250) — matrix operand order repaired and checked by reading, not compiled, not run on hardware
//
#include <hip/hip_runtime.h>
#include <hip/hip_bf16.h>
#include <math.h>
#include <stdint.h>

// ---------------- problem constants ----------------
#define E_EDGES  256000
#define NATOMS   16000
#define NF       32
#define LATD     256
#define INVD     64

typedef __bf16 bf16_t;
typedef __attribute__((ext_vector_type(16))) __bf16 v16bf;
typedef __attribute__((ext_vector_type(8)))  float  v8f;

union Frag16 { uint4 q[2]; v16bf v; };
union DFrag  { v8f v; float f[8]; };

// bf16 weight arena offsets (in bf16 elements), 1/sqrt(K) folded in at convert time
#define OFF_W1   0        // 64x256
#define OFF_W2   16384    // 256x256
#define OFF_WE0  81920    // 256x128
#define OFF_WL1A 114688   // 288x256
#define OFF_WL1B 188416   // 256x256
#define OFF_WE1  253952   // 256x64
#define OFF_WF1  270336   // 288x256
#define OFF_WF2  344064   // 256x256
#define WB_TOTAL 409600

__device__ __forceinline__ v8f wmma_bf16(v16bf a, v16bf b, v8f c) {
  return __builtin_amdgcn_wmma_f32_16x16x32_bf16(false, a, false, b, (short)0, c,
                                                 false, false);
}

// B/A fragment: 16 contiguous bf16 (32B) per lane, pre-tiled in global memory
__device__ __forceinline__ v16bf load_frag_g(const bf16_t* p) {
  Frag16 f;
  f.q[0] = *(const uint4*)(p);
  f.q[1] = *(const uint4*)(p + 8);
  return f.v;
}

// B fragment from LDS-staged panel: 32 contiguous bytes per lane
__device__ __forceinline__ v16bf b_frag_lds(const char* p) {
  Frag16 f;
  f.q[0] = *(const uint4*)(p);
  f.q[1] = *(const uint4*)(p + 16);
  return f.v;
}

// A fragment from LDS row-major [16][K] bf16 (stride_b bytes per row).
// Lane layout (16-bit A, 16x32): lane<16 holds K {0..7,16..23}, lane>=16 holds {8..15,24..31}.
__device__ __forceinline__ v16bf a_frag_lds(const char* base, int m, int stride_b,
                                            int kt, int half) {
  const char* p = base + m * stride_b + kt * 64 + half * 16;
  Frag16 f;
  f.q[0] = *(const uint4*)(p);
  f.q[1] = *(const uint4*)(p + 32);
  return f.v;
}

// A fragment converted on the fly from an f32 row (edge_inv)
__device__ __forceinline__ v16bf a_frag_f32(const float* rowp, int kbase, int half) {
  Frag16 f;
#pragma unroll
  for (int j = 0; j < 8; ++j) {
    f.v[j]     = (bf16_t)rowp[kbase + half * 8 + j];
    f.v[j + 8] = (bf16_t)rowp[kbase + 16 + half * 8 + j];
  }
  return f.v;
}

// index of element (m,k) inside an A-fragment-tiled [K/32][32 lanes][16] buffer
__device__ __forceinline__ int a_tiled_idx(int m, int k) {
  int kt = k >> 5, r = k & 31;
  int half = (r >> 3) & 1;
  int idx  = ((r >> 4) << 3) | (r & 7);
  return ((kt << 5) + (half << 4) + m) * 16 + idx;
}

__device__ __forceinline__ float silu_f(float x) { return x / (1.0f + __expf(-x)); }

// async global -> LDS staging of a 16KB B panel (128 threads x 8 x 16B)
__device__ __forceinline__ void stage_panel_async(const bf16_t* gsrc,
                                                  uint32_t lds_base, int tid) {
  uint64_t gb = (uint64_t)(uintptr_t)gsrc;
#pragma unroll
  for (int j = 0; j < 8; ++j) {
    uint32_t off   = (uint32_t)(tid * 16 + j * 2048);
    uint32_t laddr = lds_base + off;
    asm volatile("global_load_async_to_lds_b128 %0, %1, %2 offset:0"
                 :: "v"(laddr), "v"(off), "s"(gb) : "memory");
  }
}

__device__ __forceinline__ void wait_async_and_sync() {
  asm volatile("s_wait_asynccnt 0" ::: "memory");
  __syncthreads();
}

// ---------------- weight convert: f32 row-major -> bf16 B-fragment tiles ----------------
__global__ void k_cvt_btile(const float* __restrict__ src, bf16_t* __restrict__ dst,
                            int K, int N, float scale) {
  int t = blockIdx.x * 256 + threadIdx.x;
  if (t >= K * N) return;
  int idx = t & 15, lane = (t >> 4) & 31, rest = t >> 9;
  int ntiles = N >> 4;
  int nt = rest % ntiles, kt = rest / ntiles;
  int n = nt * 16 + (lane & 15);
  int k = kt * 32 + (idx >> 3) * 16 + (lane >> 4) * 8 + (idx & 7);
  dst[t] = (bf16_t)(src[k * N + n] * scale);
}

__global__ void k_zero(float* __restrict__ p, int n) {
  int t = blockIdx.x * 256 + threadIdx.x;
  if (t < n) p[t] = 0.0f;
}

// ---------------- stage 1: edge MLP -> lat, wts, feat, env_e, scatter ----------------
__global__ __launch_bounds__(128) void k_stage1(
    const float* __restrict__ edge_inv, const float* __restrict__ edge_attr,
    const int* __restrict__ eidx, const bf16_t* __restrict__ Wb,
    float* __restrict__ latf, bf16_t* __restrict__ latb,
    float* __restrict__ feat, float* __restrict__ enve, float* __restrict__ acc0) {
  __shared__ __align__(16) char sh_h[4][16 * 512];  // per-wave 16x256 bf16 (reused as f32 16x128)
  __shared__ __align__(16) char sh_panel[16384];    // staged 32x256 bf16 B panel

  const int tid = threadIdx.x;
  const int wid = tid >> 5, lane = tid & 31;
  const int g  = blockIdx.x * 4 + wid;
  const int e0 = g * 16;
  const int m = lane & 15, half = lane >> 4, cl = lane & 15;
  char* shw = sh_h[wid];
  const uint32_t panel_base = (uint32_t)(uintptr_t)(void*)sh_panel;

  __builtin_prefetch(edge_inv + (size_t)e0 * INVD, 0, 1);
  __builtin_prefetch(edge_attr + (size_t)e0 * 4, 0, 1);

  // GEMM1: h = silu(edge_inv @ W1)  (16x256, K=64)
  const float* rowp = edge_inv + (size_t)(e0 + m) * INVD;
  v16bf aI0 = a_frag_f32(rowp, 0, half);
  v16bf aI1 = a_frag_f32(rowp, 32, half);
#pragma unroll
  for (int nt = 0; nt < 16; ++nt) {
    v8f acc = (v8f)0.0f;
    acc = wmma_bf16(aI0, load_frag_g(Wb + OFF_W1 + ((size_t)(nt) * 32 + lane) * 16), acc);
    acc = wmma_bf16(aI1, load_frag_g(Wb + OFF_W1 + ((size_t)(16 + nt) * 32 + lane) * 16), acc);
    DFrag d; d.v = acc;
#pragma unroll
    for (int i = 0; i < 8; ++i) {
      int mm = i + 8 * half, n = nt * 16 + cl;
      *(bf16_t*)(shw + mm * 512 + n * 2) = (bf16_t)silu_f(d.f[i]);
    }
  }
  __syncthreads();

  // GEMM2: lat = h @ W2  (K=256), B panels async-staged into LDS
  v8f acc2[16];
#pragma unroll
  for (int nt = 0; nt < 16; ++nt) acc2[nt] = (v8f)0.0f;
  for (int kt = 0; kt < 8; ++kt) {
    stage_panel_async(Wb + OFF_W2 + kt * 8192, panel_base, tid);
    wait_async_and_sync();
    v16bf af = a_frag_lds(shw, m, 512, kt, half);
#pragma unroll
    for (int nt = 0; nt < 16; ++nt)
      acc2[nt] = wmma_bf16(af, b_frag_lds((const char*)sh_panel + (nt * 32 + lane) * 32),
                           acc2[nt]);
    __syncthreads();
  }

  // write lat: f32 running copy (d_out), bf16 A-tiled (global), bf16 rows (LDS)
  bf16_t* latb_g = latb + (size_t)g * 4096;
#pragma unroll
  for (int nt = 0; nt < 16; ++nt) {
    DFrag d; d.v = acc2[nt];
#pragma unroll
    for (int i = 0; i < 8; ++i) {
      int mm = i + 8 * half, n = nt * 16 + cl;
      float v = d.f[i];
      latf[(size_t)(e0 + mm) * LATD + n] = v;
      latb_g[a_tiled_idx(mm, n)] = (bf16_t)v;
      *(bf16_t*)(shw + mm * 512 + n * 2) = (bf16_t)v;
    }
  }

  // env0: wts = lat @ W_env0 (K=256, N=128), B direct (L2-resident weights)
  v8f accw[8];
#pragma unroll
  for (int nt = 0; nt < 8; ++nt) accw[nt] = (v8f)0.0f;
#pragma unroll
  for (int kt = 0; kt < 8; ++kt) {
    v16bf af = a_frag_lds(shw, m, 512, kt, half);
#pragma unroll
    for (int nt = 0; nt < 8; ++nt)
      accw[nt] = wmma_bf16(af,
          load_frag_g(Wb + OFF_WE0 + ((size_t)(kt * 8 + nt) * 32 + lane) * 16), accw[nt]);
  }
  float* wts = (float*)shw;  // reuse as 16x128 f32
#pragma unroll
  for (int nt = 0; nt < 8; ++nt) {
    DFrag d; d.v = accw[nt];
#pragma unroll
    for (int i = 0; i < 8; ++i) wts[(i + 8 * half) * 128 + nt * 16 + cl] = d.f[i];
  }

  // scalar: feat / env_e / scatter-add into per-atom accumulator (L2-resident)
  const float envc = 0.2581988897471611f;  // 1/sqrt(15)
#pragma unroll
  for (int j = 0; j < 16; ++j) {
    int p = j * 32 + lane, el = p >> 5, f = p & 31, e = e0 + el;
    float w0 = wts[el * 128 + 2 * f], w1 = wts[el * 128 + 2 * f + 1];
    float u0 = wts[el * 128 + 64 + 2 * f], u1 = wts[el * 128 + 64 + 2 * f + 1];
    float4 at = *(const float4*)(edge_attr + (size_t)e * 4);
    float4 ft = make_float4(w0 * at.x, w1 * at.y, w1 * at.z, w1 * at.w);
    float4 ev = make_float4(envc * u0 * at.x, envc * u1 * at.y,
                            envc * u1 * at.z, envc * u1 * at.w);
    *(float4*)(feat + ((size_t)e * 32 + f) * 4) = ft;
    *(float4*)(enve + ((size_t)e * 32 + f) * 4) = ev;
    int c = eidx[e];
    float* ap = acc0 + ((size_t)c * 32 + f) * 4;
    atomicAdd(ap + 0, ev.x); atomicAdd(ap + 1, ev.y);
    atomicAdd(ap + 2, ev.z); atomicAdd(ap + 3, ev.w);
  }
}

// ---------------- stage 2: tp0, latent MLP #2, env1 scatter ----------------
__global__ __launch_bounds__(128) void k_stage2(
    const float* __restrict__ edge_attr, const int* __restrict__ eidx,
    const bf16_t* __restrict__ Wb, const float* __restrict__ w_tp0,
    float* __restrict__ latf, bf16_t* __restrict__ latb,
    float* __restrict__ feat, float* __restrict__ enve,
    const float* __restrict__ acc0, float* __restrict__ acc1) {
  __shared__ __align__(16) char sh_h[4][16 * 512];
  __shared__ __align__(16) char sh_panel[16384];
  __shared__ __align__(16) char sh_scal[4][16 * 64];  // per-wave 16x32 bf16

  const int tid = threadIdx.x;
  const int wid = tid >> 5, lane = tid & 31;
  const int g  = blockIdx.x * 4 + wid;
  const int e0 = g * 16;
  const int m = lane & 15, half = lane >> 4, cl = lane & 15;
  char* shw = sh_h[wid];
  char* shs = sh_scal[wid];
  const uint32_t panel_base = (uint32_t)(uintptr_t)(void*)sh_panel;
  const float RS2 = 0.7071067811865476f;
  const float IS3 = 0.5773502691896258f;

  // scalar: env gather, tensor product 0, scal -> LDS bf16
#pragma unroll
  for (int j = 0; j < 16; ++j) {
    int p = j * 32 + lane, el = p >> 5, f = p & 31, e = e0 + el;
    int c = eidx[e];
    float4 ft = *(const float4*)(feat + ((size_t)e * 32 + f) * 4);
    float4 ee = *(const float4*)(enve + ((size_t)e * 32 + f) * 4);
    const float* ag = acc0 + ((size_t)c * 32 + f) * 4;
    float n0 = ag[0] - ee.x, n1 = ag[1] - ee.y, n2 = ag[2] - ee.z, n3 = ag[3] - ee.w;
    float wt0 = w_tp0[f], wt1 = w_tp0[32 + f], wt2 = w_tp0[64 + f], wt3 = w_tp0[96 + f];
    float dotv = ft.y * n1 + ft.z * n2 + ft.w * n3;
    float o0 = wt0 * ft.x * n0 + wt1 * dotv * IS3;
    float o1 = wt2 * ft.x * n1 + wt3 * ft.y * n0;
    float o2 = wt2 * ft.x * n2 + wt3 * ft.z * n0;
    float o3 = wt2 * ft.x * n3 + wt3 * ft.w * n0;
    *(float4*)(feat + ((size_t)e * 32 + f) * 4) = make_float4(o0, o1, o2, o3);
    *(bf16_t*)(shs + el * 64 + f * 2) = (bf16_t)o0;
  }
  __syncthreads();

  // l1a: h = silu([lat | scal] @ W_l1a)  (K=288)
  v8f acc[16];
#pragma unroll
  for (int nt = 0; nt < 16; ++nt) acc[nt] = (v8f)0.0f;
  for (int kt = 0; kt < 9; ++kt) {
    stage_panel_async(Wb + OFF_WL1A + kt * 8192, panel_base, tid);
    wait_async_and_sync();
    v16bf af = (kt < 8)
        ? load_frag_g(latb + (size_t)g * 4096 + ((size_t)kt * 32 + lane) * 16)
        : a_frag_lds(shs, m, 64, 0, half);
#pragma unroll
    for (int nt = 0; nt < 16; ++nt)
      acc[nt] = wmma_bf16(af, b_frag_lds((const char*)sh_panel + (nt * 32 + lane) * 32),
                          acc[nt]);
    __syncthreads();
  }
#pragma unroll
  for (int nt = 0; nt < 16; ++nt) {
    DFrag d; d.v = acc[nt];
#pragma unroll
    for (int i = 0; i < 8; ++i) {
      int mm = i + 8 * half, n = nt * 16 + cl;
      *(bf16_t*)(shw + mm * 512 + n * 2) = (bf16_t)silu_f(d.f[i]);
    }
  }
  __syncthreads();

  // l1b: new_lat = h @ W_l1b  (K=256)
  v8f acc2[16];
#pragma unroll
  for (int nt = 0; nt < 16; ++nt) acc2[nt] = (v8f)0.0f;
  for (int kt = 0; kt < 8; ++kt) {
    stage_panel_async(Wb + OFF_WL1B + kt * 8192, panel_base, tid);
    wait_async_and_sync();
    v16bf af = a_frag_lds(shw, m, 512, kt, half);
#pragma unroll
    for (int nt = 0; nt < 16; ++nt)
      acc2[nt] = wmma_bf16(af, b_frag_lds((const char*)sh_panel + (nt * 32 + lane) * 32),
                           acc2[nt]);
    __syncthreads();
  }

  // residual update lat = (lat + new_lat) * rsqrt(2); refresh f32/bf16/LDS copies
  bf16_t* latb_g = latb + (size_t)g * 4096;
#pragma unroll
  for (int nt = 0; nt < 16; ++nt) {
    DFrag d; d.v = acc2[nt];
#pragma unroll
    for (int i = 0; i < 8; ++i) {
      int mm = i + 8 * half, n = nt * 16 + cl;
      size_t o = (size_t)(e0 + mm) * LATD + n;
      float v = (latf[o] + d.f[i]) * RS2;
      latf[o] = v;
      latb_g[a_tiled_idx(mm, n)] = (bf16_t)v;
      *(bf16_t*)(shw + mm * 512 + n * 2) = (bf16_t)v;
    }
  }

  // env1: env_w = lat @ W_env1 (K=256, N=64)
  v8f accw[4];
#pragma unroll
  for (int nt = 0; nt < 4; ++nt) accw[nt] = (v8f)0.0f;
#pragma unroll
  for (int kt = 0; kt < 8; ++kt) {
    v16bf af = a_frag_lds(shw, m, 512, kt, half);
#pragma unroll
    for (int nt = 0; nt < 4; ++nt)
      accw[nt] = wmma_bf16(af,
          load_frag_g(Wb + OFF_WE1 + ((size_t)(kt * 4 + nt) * 32 + lane) * 16), accw[nt]);
  }
  float* wts = (float*)shw;  // reuse as 16x64 f32
#pragma unroll
  for (int nt = 0; nt < 4; ++nt) {
    DFrag d; d.v = accw[nt];
#pragma unroll
    for (int i = 0; i < 8; ++i) wts[(i + 8 * half) * 64 + nt * 16 + cl] = d.f[i];
  }

  // scalar: env_e2 + scatter-add into acc1
  const float envc = 0.2581988897471611f;
#pragma unroll
  for (int j = 0; j < 16; ++j) {
    int p = j * 32 + lane, el = p >> 5, f = p & 31, e = e0 + el;
    float u0 = wts[el * 64 + 2 * f], u1 = wts[el * 64 + 2 * f + 1];
    float4 at = *(const float4*)(edge_attr + (size_t)e * 4);
    float4 ev = make_float4(envc * u0 * at.x, envc * u1 * at.y,
                            envc * u1 * at.z, envc * u1 * at.w);
    *(float4*)(enve + ((size_t)e * 32 + f) * 4) = ev;
    int c = eidx[e];
    float* ap = acc1 + ((size_t)c * 32 + f) * 4;
    atomicAdd(ap + 0, ev.x); atomicAdd(ap + 1, ev.y);
    atomicAdd(ap + 2, ev.z); atomicAdd(ap + 3, ev.w);
  }
}

// ---------------- stage 3: tp1, final latent MLP ----------------
__global__ __launch_bounds__(128) void k_stage3(
    const int* __restrict__ eidx, const bf16_t* __restrict__ Wb,
    const float* __restrict__ w_tp1, float* __restrict__ latf,
    const bf16_t* __restrict__ latb, const float* __restrict__ feat,
    const float* __restrict__ enve, const float* __restrict__ acc1) {
  __shared__ __align__(16) char sh_h[4][16 * 512];
  __shared__ __align__(16) char sh_panel[16384];
  __shared__ __align__(16) char sh_scal[4][16 * 64];

  const int tid = threadIdx.x;
  const int wid = tid >> 5, lane = tid & 31;
  const int g  = blockIdx.x * 4 + wid;
  const int e0 = g * 16;
  const int m = lane & 15, half = lane >> 4, cl = lane & 15;
  char* shw = sh_h[wid];
  char* shs = sh_scal[wid];
  const uint32_t panel_base = (uint32_t)(uintptr_t)(void*)sh_panel;
  const float RS2 = 0.7071067811865476f;
  const float IS3 = 0.5773502691896258f;

  // scalar: env gather #2, tensor product 1 -> scal in LDS bf16
#pragma unroll
  for (int j = 0; j < 16; ++j) {
    int p = j * 32 + lane, el = p >> 5, f = p & 31, e = e0 + el;
    int c = eidx[e];
    float4 ft = *(const float4*)(feat + ((size_t)e * 32 + f) * 4);
    float4 ee = *(const float4*)(enve + ((size_t)e * 32 + f) * 4);
    const float* ag = acc1 + ((size_t)c * 32 + f) * 4;
    float n0 = ag[0] - ee.x, n1 = ag[1] - ee.y, n2 = ag[2] - ee.z, n3 = ag[3] - ee.w;
    float wt0 = w_tp1[f], wt1 = w_tp1[32 + f];
    float dotv = ft.y * n1 + ft.z * n2 + ft.w * n3;
    float o0 = wt0 * ft.x * n0 + wt1 * dotv * IS3;
    *(bf16_t*)(shs + el * 64 + f * 2) = (bf16_t)o0;
  }
  __syncthreads();

  // f1: h = silu([lat | scal2] @ W_f1)  (K=288)
  v8f acc[16];
#pragma unroll
  for (int nt = 0; nt < 16; ++nt) acc[nt] = (v8f)0.0f;
  for (int kt = 0; kt < 9; ++kt) {
    stage_panel_async(Wb + OFF_WF1 + kt * 8192, panel_base, tid);
    wait_async_and_sync();
    v16bf af = (kt < 8)
        ? load_frag_g(latb + (size_t)g * 4096 + ((size_t)kt * 32 + lane) * 16)
        : a_frag_lds(shs, m, 64, 0, half);
#pragma unroll
    for (int nt = 0; nt < 16; ++nt)
      acc[nt] = wmma_bf16(af, b_frag_lds((const char*)sh_panel + (nt * 32 + lane) * 32),
                          acc[nt]);
    __syncthreads();
  }
#pragma unroll
  for (int nt = 0; nt < 16; ++nt) {
    DFrag d; d.v = acc[nt];
#pragma unroll
    for (int i = 0; i < 8; ++i) {
      int mm = i + 8 * half, n = nt * 16 + cl;
      *(bf16_t*)(shw + mm * 512 + n * 2) = (bf16_t)silu_f(d.f[i]);
    }
  }
  __syncthreads();

  // f2: new_lat = h @ W_f2 (K=256); final residual to d_out
  v8f acc2[16];
#pragma unroll
  for (int nt = 0; nt < 16; ++nt) acc2[nt] = (v8f)0.0f;
  for (int kt = 0; kt < 8; ++kt) {
    stage_panel_async(Wb + OFF_WF2 + kt * 8192, panel_base, tid);
    wait_async_and_sync();
    v16bf af = a_frag_lds(shw, m, 512, kt, half);
#pragma unroll
    for (int nt = 0; nt < 16; ++nt)
      acc2[nt] = wmma_bf16(af, b_frag_lds((const char*)sh_panel + (nt * 32 + lane) * 32),
                           acc2[nt]);
    __syncthreads();
  }
#pragma unroll
  for (int nt = 0; nt < 16; ++nt) {
    DFrag d; d.v = acc2[nt];
#pragma unroll
    for (int i = 0; i < 8; ++i) {
      int mm = i + 8 * half, n = nt * 16 + cl;
      size_t o = (size_t)(e0 + mm) * LATD + n;
      latf[o] = (latf[o] + d.f[i]) * RS2;
    }
  }
}

// ---------------- host launcher ----------------
extern "C" void kernel_launch(void* const* d_in, const int* in_sizes, int n_in,
                              void* d_out, int out_size, void* d_ws, size_t ws_size,
                              hipStream_t stream) {
  const int*   eidx      = (const int*)d_in[0];    // (2, E) int32, row 0 = edge_center
  const float* edge_attr = (const float*)d_in[1];  // (E, 4)
  const float* edge_inv  = (const float*)d_in[2];  // (E, 64)
  const float* W_tb1  = (const float*)d_in[3];
  const float* W_tb2  = (const float*)d_in[4];
  const float* W_env0 = (const float*)d_in[5];
  const float* W_l1a  = (const float*)d_in[6];
  const float* W_l1b  = (const float*)d_in[7];
  const float* W_env1 = (const float*)d_in[8];
  const float* W_f1   = (const float*)d_in[9];
  const float* W_f2   = (const float*)d_in[10];
  const float* w_tp0  = (const float*)d_in[11];
  const float* w_tp1  = (const float*)d_in[12];

  float* latf = (float*)d_out;  // running latent (E, 256) f32

  char* ws = (char*)d_ws;
  bf16_t* latb = (bf16_t*)ws;                        // E*256 bf16, A-fragment tiled
  float*  feat = (float*)(ws + 131072000ull);        // E*32*4 f32
  float*  enve = (float*)(ws + 262144000ull);        // E*32*4 f32
  float*  acc0 = (float*)(ws + 393216000ull);        // NATOMS*128 f32
  float*  acc1 = (float*)(ws + 401408000ull);        // NATOMS*128 f32
  bf16_t* Wb   = (bf16_t*)(ws + 409600000ull);       // all bf16 weights, B-tiled

  // one-time-per-call weight conversion (tiny): fold 1/sqrt(K) scales
  auto cvt = [&](const float* s, int off, int K, int N, float sc) {
    int tot = K * N;
    k_cvt_btile<<<(tot + 255) / 256, 256, 0, stream>>>(s, Wb + off, K, N, sc);
  };
  const float s288 = 1.0f / sqrtf(288.0f);
  cvt(W_tb1,  OFF_W1,   64,  256, 0.125f);   // 1/sqrt(64)
  cvt(W_tb2,  OFF_W2,   256, 256, 0.0625f);  // 1/sqrt(256)
  cvt(W_env0, OFF_WE0,  256, 128, 0.0625f);
  cvt(W_l1a,  OFF_WL1A, 288, 256, s288);
  cvt(W_l1b,  OFF_WL1B, 256, 256, 0.0625f);
  cvt(W_env1, OFF_WE1,  256, 64,  0.0625f);
  cvt(W_f1,   OFF_WF1,  288, 256, s288);
  cvt(W_f2,   OFF_WF2,  256, 256, 0.0625f);

  // zero both atom accumulators (contiguous in ws)
  {
    int n = 2 * NATOMS * 128;
    k_zero<<<(n + 255) / 256, 256, 0, stream>>>(acc0, n);
  }

  const int nblk = E_EDGES / 64;  // 4 waves x 16 edges per block
  k_stage1<<<nblk, 128, 0, stream>>>(edge_inv, edge_attr, eidx, Wb,
                                     latf, latb, feat, enve, acc0);
  k_stage2<<<nblk, 128, 0, stream>>>(edge_attr, eidx, Wb, w_tp0,
                                     latf, latb, feat, enve, acc0, acc1);
  k_stage3<<<nblk, 128, 0, stream>>>(eidx, Wb, w_tp1, latf, latb, feat, enve, acc1);
}
